// KnowledgeAwareAttention_28467043238223
// MI455X (gfx1250) — matrix operand order repaired and checked by reading, not compiled, not run on hardware
//
#include <hip/hip_runtime.h>

// ---------------------------------------------------------------------------
// KnowledgeAwareAttention for MI455X (gfx1250, wave32, WMMA bf16 16x16x32)
//
// scores = (Q@K^T + Q@KG^T)*s == (Q @ (K+KG)^T)*s   -> fold KG into K ("KS")
// All GEMMs via v_wmma_f32_16x16x32_bf16 (f32 accumulate), softmax in f32.
// Per-wave tile: 32(M) x 64(N): 2 A-frags x 4 B-frags -> 8 WMMA per K-chunk.
// ---------------------------------------------------------------------------

#define EMB   768
#define SEQ   2048
#define NB    2
#define NH    12
#define DH    64
#define BH    (NB*NH)          // 24
#define BS    (NB*SEQ)         // 4096
#define ESQ   (EMB*EMB)        // 589824

typedef __attribute__((ext_vector_type(16))) __bf16  v16bf;
typedef __attribute__((ext_vector_type(8)))  __bf16  v8bf;
typedef __attribute__((ext_vector_type(4)))  __bf16  v4bf;
typedef __attribute__((ext_vector_type(8)))  float   v8f;

union AFrag { v16bf v; __bf16 e[16]; v8bf h[2]; };

static __device__ __forceinline__ v8f wmma_bf16(v16bf a, v16bf b, v8f c) {
  // (neg_a, A, neg_b, B, c_mod, C, reuse_a, reuse_b)
  return __builtin_amdgcn_wmma_f32_16x16x32_bf16(false, a, false, b,
                                                 (short)0, c, false, false);
}

// convert two float4 chunks into 8 consecutive A-frag elements (native cvt)
static __device__ __forceinline__ void cvt8(AFrag& a, int base, float4 x, float4 y) {
  a.e[base + 0] = (__bf16)x.x; a.e[base + 1] = (__bf16)x.y;
  a.e[base + 2] = (__bf16)x.z; a.e[base + 3] = (__bf16)x.w;
  a.e[base + 4] = (__bf16)y.x; a.e[base + 5] = (__bf16)y.y;
  a.e[base + 6] = (__bf16)y.z; a.e[base + 7] = (__bf16)y.w;
}

// ---------------------------------------------------------------- cvt f32->bf16
__global__ __launch_bounds__(256) void cvt_kernel(const float* __restrict__ src,
                                                  __bf16* __restrict__ dst, int n4) {
  int i = blockIdx.x * blockDim.x + threadIdx.x;
  if (i < n4) {
    float4 t = ((const float4*)src)[i];
    v4bf o; o[0] = (__bf16)t.x; o[1] = (__bf16)t.y; o[2] = (__bf16)t.z; o[3] = (__bf16)t.w;
    ((v4bf*)dst)[i] = o;
  }
}

// ------------------------------------------------- projection GEMM (dual input)
// OUT = bf16( X1@W1 + (X2?X2@W2:0) + b1 (+b2) ), X f32 [4096,768], W bf16 [768,768]
// store_mode 0: row-major [4096,768]; 1: head-transposed KST [bh][d][s]
__global__ __launch_bounds__(256) void proj_kernel(
    const float* __restrict__ X1, const __bf16* __restrict__ W1,
    const float* __restrict__ X2, const __bf16* __restrict__ W2,
    const float* __restrict__ b1, const float* __restrict__ b2,
    __bf16* __restrict__ OUT, int store_mode) {
  const int wave  = blockIdx.x * (blockDim.x >> 5) + (threadIdx.x >> 5);
  const int lane  = threadIdx.x & 31;
  const int m     = lane & 15, hi = lane >> 4;
  const int mtile = wave / (EMB / 64);          // 0..127 (32 rows each)
  const int ngrp  = wave % (EMB / 64);          // 0..11  (64 cols each)
  const int row0  = mtile * 32;
  const int n0b   = ngrp * 64;

  v8f c[8] = {};
  for (int pass = 0; pass < 2; ++pass) {
    const float*  X = pass ? X2 : X1;
    const __bf16* W = pass ? W2 : W1;
    if (!X) break;
    for (int k0 = 0; k0 < EMB; k0 += 32) {
      AFrag a0, a1;
      {
        const float4* xa = (const float4*)(X + (long)(row0 + m) * EMB + k0 + hi * 8);
        cvt8(a0, 0, xa[0], xa[1]); cvt8(a0, 8, xa[4], xa[5]);
        const float4* xb = (const float4*)(X + (long)(row0 + 16 + m) * EMB + k0 + hi * 8);
        cvt8(a1, 0, xb[0], xb[1]); cvt8(a1, 8, xb[4], xb[5]);
      }
      const int klane = k0 + hi * 8;
      v16bf bf[4];
#pragma unroll
      for (int j = 0; j < 4; ++j)
        for (int s = 0; s < 16; ++s)
          bf[j][s] = W[(long)(klane + (s & 7) + (s >> 3) * 16) * EMB + n0b + j * 16 + m];
#pragma unroll
      for (int j = 0; j < 4; ++j) {
        c[j]     = wmma_bf16(a0.v, bf[j], c[j]);
        c[4 + j] = wmma_bf16(a1.v, bf[j], c[4 + j]);
      }
    }
  }
#pragma unroll
  for (int j = 0; j < 4; ++j) {
    const int col  = n0b + j * 16 + m;
    const float bias = b1[col] + (b2 ? b2[col] : 0.0f);
#pragma unroll
    for (int t = 0; t < 2; ++t) {
#pragma unroll
      for (int v = 0; v < 8; ++v) {
        const int r = row0 + t * 16 + hi * 8 + v;
        const __bf16 val = (__bf16)(c[t * 4 + j][v] + bias);
        if (store_mode == 0) {
          OUT[(long)r * EMB + col] = val;
        } else {  // KST[(b*NH+h)][d][s]
          const int b = r >> 11, s = r & (SEQ - 1);
          const int h = col >> 6, d = col & 63;
          OUT[(((long)(b * NH + h)) * DH + d) * SEQ + s] = val;
        }
      }
    }
  }
}

// ----------------------------------------------- scores = scale*(Q@KS^T), mask
__global__ __launch_bounds__(256) void scores_kernel(
    const __bf16* __restrict__ Qb, const __bf16* __restrict__ KST,
    const int* __restrict__ mask, float* __restrict__ attnW) {
  const int wave = blockIdx.x * (blockDim.x >> 5) + (threadIdx.x >> 5);
  const int lane = threadIdx.x & 31;
  const int m = lane & 15, hi = lane >> 4;
  const int kg = wave & 31;            // 32 groups of 64 k-cols
  const int qt = (wave >> 5) & 63;     // q tile (32 rows)
  const int bh = wave >> 11;           // 0..23
  const int b = bh / NH, h = bh % NH;
  const int q0 = qt * 32, k0b = kg * 64;

  v8f c[8] = {};
#pragma unroll
  for (int d0 = 0; d0 < DH; d0 += 32) {
    AFrag a0, a1;
    {
      const v8bf* qa = (const v8bf*)(Qb + ((long)b * SEQ + q0 + m) * EMB + h * DH + d0 + hi * 8);
      a0.h[0] = qa[0]; a0.h[1] = qa[2];
      const v8bf* qb = (const v8bf*)(Qb + ((long)b * SEQ + q0 + 16 + m) * EMB + h * DH + d0 + hi * 8);
      a1.h[0] = qb[0]; a1.h[1] = qb[2];
    }
    const int dlane = d0 + hi * 8;
    v16bf bf[4];
#pragma unroll
    for (int j = 0; j < 4; ++j)
      for (int s = 0; s < 16; ++s)
        bf[j][s] = KST[((long)bh * DH + dlane + (s & 7) + (s >> 3) * 16) * SEQ + k0b + j * 16 + m];
#pragma unroll
    for (int j = 0; j < 4; ++j) {
      c[j]     = wmma_bf16(a0.v, bf[j], c[j]);
      c[4 + j] = wmma_bf16(a1.v, bf[j], c[4 + j]);
    }
  }
  const float scale = 0.125f;  // 1/sqrt(64)
  const int* mb = mask + (long)b * SEQ * SEQ;
#pragma unroll
  for (int j = 0; j < 4; ++j) {
    const int k = k0b + j * 16 + m;
#pragma unroll
    for (int t = 0; t < 2; ++t) {
#pragma unroll
      for (int v = 0; v < 8; ++v) {
        const int q = q0 + t * 16 + hi * 8 + v;
        float s = c[t * 4 + j][v] * scale;
        if (mb[(long)q * SEQ + k] == 0) s = -1e9f;
        attnW[((long)bh * SEQ + q) * SEQ + k] = s;
      }
    }
  }
}

// -------------------------------------------- per-row softmax stats (max, sum)
__global__ __launch_bounds__(256) void stats_kernel(const float* __restrict__ attnW,
                                                    float2* __restrict__ stats) {
  const int wave = blockIdx.x * (blockDim.x >> 5) + (threadIdx.x >> 5);  // row id
  const int lane = threadIdx.x & 31;
  const long base = (long)wave * SEQ;
  float mx = -3.4e38f;
  for (int i = lane * 4; i < SEQ; i += 128) {
    float4 t = *(const float4*)(attnW + base + i);
    mx = fmaxf(mx, fmaxf(fmaxf(t.x, t.y), fmaxf(t.z, t.w)));
  }
#pragma unroll
  for (int off = 16; off; off >>= 1) mx = fmaxf(mx, __shfl_xor(mx, off, 32));
  float sum = 0.0f;
  for (int i = lane * 4; i < SEQ; i += 128) {
    float4 t = *(const float4*)(attnW + base + i);
    sum += __expf(t.x - mx) + __expf(t.y - mx) + __expf(t.z - mx) + __expf(t.w - mx);
  }
#pragma unroll
  for (int off = 16; off; off >>= 1) sum += __shfl_xor(sum, off, 32);
  if (lane == 0) stats[wave] = make_float2(mx, sum);
}

// ------------------- normalize scores in place (-> attn_weights) and attn @ V
__global__ __launch_bounds__(256) void normav_kernel(
    float* __restrict__ attnW, const __bf16* __restrict__ Vb,
    const float2* __restrict__ stats, __bf16* __restrict__ AOb) {
  const int wave = blockIdx.x * (blockDim.x >> 5) + (threadIdx.x >> 5);  // 0..1535
  const int lane = threadIdx.x & 31;
  const int m = lane & 15, hi = lane >> 4;
  const int qt = wave & 63;
  const int bh = wave >> 6;
  const int b = bh / NH, h = bh % NH;
  const int q0 = qt * 32;
  const float2 st0 = stats[bh * SEQ + q0 + m];
  const float2 st1 = stats[bh * SEQ + q0 + 16 + m];
  const float mx0 = st0.x, inv0 = 1.0f / st0.y;
  const float mx1 = st1.x, inv1 = 1.0f / st1.y;
  float* wrow0 = attnW + ((long)bh * SEQ + q0 + m) * SEQ;
  float* wrow1 = attnW + ((long)bh * SEQ + q0 + 16 + m) * SEQ;

  v8f c[8] = {};
  for (int k0 = 0; k0 < SEQ; k0 += 32) {
    AFrag a0, a1;
    {
      float4* p = (float4*)(wrow0 + k0 + hi * 8);
      float4 r0 = p[0], r1 = p[1], r2 = p[4], r3 = p[5];
      r0.x = __expf(r0.x - mx0) * inv0; r0.y = __expf(r0.y - mx0) * inv0;
      r0.z = __expf(r0.z - mx0) * inv0; r0.w = __expf(r0.w - mx0) * inv0;
      r1.x = __expf(r1.x - mx0) * inv0; r1.y = __expf(r1.y - mx0) * inv0;
      r1.z = __expf(r1.z - mx0) * inv0; r1.w = __expf(r1.w - mx0) * inv0;
      r2.x = __expf(r2.x - mx0) * inv0; r2.y = __expf(r2.y - mx0) * inv0;
      r2.z = __expf(r2.z - mx0) * inv0; r2.w = __expf(r2.w - mx0) * inv0;
      r3.x = __expf(r3.x - mx0) * inv0; r3.y = __expf(r3.y - mx0) * inv0;
      r3.z = __expf(r3.z - mx0) * inv0; r3.w = __expf(r3.w - mx0) * inv0;
      p[0] = r0; p[1] = r1; p[4] = r2; p[5] = r3;  // materialize attn_weights
      cvt8(a0, 0, r0, r1); cvt8(a0, 8, r2, r3);
    }
    {
      float4* p = (float4*)(wrow1 + k0 + hi * 8);
      float4 r0 = p[0], r1 = p[1], r2 = p[4], r3 = p[5];
      r0.x = __expf(r0.x - mx1) * inv1; r0.y = __expf(r0.y - mx1) * inv1;
      r0.z = __expf(r0.z - mx1) * inv1; r0.w = __expf(r0.w - mx1) * inv1;
      r1.x = __expf(r1.x - mx1) * inv1; r1.y = __expf(r1.y - mx1) * inv1;
      r1.z = __expf(r1.z - mx1) * inv1; r1.w = __expf(r1.w - mx1) * inv1;
      r2.x = __expf(r2.x - mx1) * inv1; r2.y = __expf(r2.y - mx1) * inv1;
      r2.z = __expf(r2.z - mx1) * inv1; r2.w = __expf(r2.w - mx1) * inv1;
      r3.x = __expf(r3.x - mx1) * inv1; r3.y = __expf(r3.y - mx1) * inv1;
      r3.z = __expf(r3.z - mx1) * inv1; r3.w = __expf(r3.w - mx1) * inv1;
      p[0] = r0; p[1] = r1; p[4] = r2; p[5] = r3;
      cvt8(a1, 0, r0, r1); cvt8(a1, 8, r2, r3);
    }
    const int klane = k0 + hi * 8;
    v16bf bf[4];
#pragma unroll
    for (int j = 0; j < 4; ++j)
      for (int s = 0; s < 16; ++s)
        bf[j][s] = Vb[((long)b * SEQ + klane + (s & 7) + (s >> 3) * 16) * EMB + h * DH + j * 16 + m];
#pragma unroll
    for (int j = 0; j < 4; ++j) {
      c[j]     = wmma_bf16(a0.v, bf[j], c[j]);
      c[4 + j] = wmma_bf16(a1.v, bf[j], c[4 + j]);
    }
  }
#pragma unroll
  for (int j = 0; j < 4; ++j) {
    const int d = j * 16 + m;
#pragma unroll
    for (int t = 0; t < 2; ++t) {
#pragma unroll
      for (int v = 0; v < 8; ++v) {
        const int q = q0 + t * 16 + hi * 8 + v;
        AOb[((long)b * SEQ + q) * EMB + h * DH + d] = (__bf16)(c[t * 4 + j][v]);
      }
    }
  }
}

// --------------------------------------------------- output GEMM (f32 result)
__global__ __launch_bounds__(256) void out_kernel(
    const __bf16* __restrict__ AOb, const __bf16* __restrict__ Wo,
    const float* __restrict__ bo, float* __restrict__ OUT) {
  const int wave = blockIdx.x * (blockDim.x >> 5) + (threadIdx.x >> 5);
  const int lane = threadIdx.x & 31;
  const int m = lane & 15, hi = lane >> 4;
  const int mtile = wave / (EMB / 64);
  const int ngrp  = wave % (EMB / 64);
  const int row0 = mtile * 32;
  const int n0b  = ngrp * 64;

  v8f c[8] = {};
  for (int k0 = 0; k0 < EMB; k0 += 32) {
    AFrag a0, a1;
    {
      const v8bf* pa = (const v8bf*)(AOb + (long)(row0 + m) * EMB + k0 + hi * 8);
      a0.h[0] = pa[0]; a0.h[1] = pa[2];
      const v8bf* pb = (const v8bf*)(AOb + (long)(row0 + 16 + m) * EMB + k0 + hi * 8);
      a1.h[0] = pb[0]; a1.h[1] = pb[2];
    }
    const int klane = k0 + hi * 8;
    v16bf bf[4];
#pragma unroll
    for (int j = 0; j < 4; ++j)
      for (int s = 0; s < 16; ++s)
        bf[j][s] = Wo[(long)(klane + (s & 7) + (s >> 3) * 16) * EMB + n0b + j * 16 + m];
#pragma unroll
    for (int j = 0; j < 4; ++j) {
      c[j]     = wmma_bf16(a0.v, bf[j], c[j]);
      c[4 + j] = wmma_bf16(a1.v, bf[j], c[4 + j]);
    }
  }
#pragma unroll
  for (int j = 0; j < 4; ++j) {
    const int col = n0b + j * 16 + m;
    const float bias = bo[col];
#pragma unroll
    for (int t = 0; t < 2; ++t) {
#pragma unroll
      for (int v = 0; v < 8; ++v) {
        const int r = row0 + t * 16 + hi * 8 + v;
        OUT[(long)r * EMB + col] = c[t * 4 + j][v] + bias;
      }
    }
  }
}

// ---------------------------------------------------------------------------
extern "C" void kernel_launch(void* const* d_in, const int* in_sizes, int n_in,
                              void* d_out, int out_size, void* d_ws, size_t ws_size,
                              hipStream_t stream) {
  (void)in_sizes; (void)n_in; (void)out_size; (void)ws_size;
  const float* query = (const float*)d_in[0];
  const float* key_  = (const float*)d_in[1];
  const float* value = (const float*)d_in[2];
  const float* kginf = (const float*)d_in[3];
  const int*   mask  = (const int*)d_in[4];
  const float* Wq = (const float*)d_in[5];  const float* bq  = (const float*)d_in[6];
  const float* Wk = (const float*)d_in[7];  const float* bk  = (const float*)d_in[8];
  const float* Wv = (const float*)d_in[9];  const float* bv  = (const float*)d_in[10];
  const float* Wkg= (const float*)d_in[11]; const float* bkg = (const float*)d_in[12];
  const float* Wo = (const float*)d_in[13]; const float* bo  = (const float*)d_in[14];

  float* out   = (float*)d_out;
  float* attnW = out + (size_t)BS * EMB;            // attn_weights region of d_out

  // workspace layout (bf16 buffers + row stats), ~31.5 MB
  __bf16* wgt = (__bf16*)d_ws;                      // 5 weight matrices (bf16)
  __bf16* Wq_b  = wgt + 0 * (size_t)ESQ;
  __bf16* Wk_b  = wgt + 1 * (size_t)ESQ;
  __bf16* Wv_b  = wgt + 2 * (size_t)ESQ;
  __bf16* Wkg_b = wgt + 3 * (size_t)ESQ;
  __bf16* Wo_b  = wgt + 4 * (size_t)ESQ;
  __bf16* Qb  = wgt + 5 * (size_t)ESQ;              // [4096,768]
  __bf16* KST = Qb  + (size_t)BS * EMB;             // [24][64][2048]
  __bf16* Vb  = KST + (size_t)BS * EMB;             // [4096,768]
  __bf16* AOb = Vb  + (size_t)BS * EMB;             // [4096,768]
  float2* stats = (float2*)(AOb + (size_t)BS * EMB);// [24*2048] (max,sum)

  const dim3 blk(256);
  // 1. convert weights f32 -> bf16 (vectorized x4)
  cvt_kernel<<<dim3(ESQ / 4 / 256), blk, 0, stream>>>(Wq,  Wq_b,  ESQ / 4);
  cvt_kernel<<<dim3(ESQ / 4 / 256), blk, 0, stream>>>(Wk,  Wk_b,  ESQ / 4);
  cvt_kernel<<<dim3(ESQ / 4 / 256), blk, 0, stream>>>(Wv,  Wv_b,  ESQ / 4);
  cvt_kernel<<<dim3(ESQ / 4 / 256), blk, 0, stream>>>(Wkg, Wkg_b, ESQ / 4);
  cvt_kernel<<<dim3(ESQ / 4 / 256), blk, 0, stream>>>(Wo,  Wo_b,  ESQ / 4);

  // 2. projections (1536 waves each = 192 blocks of 8 waves)
  proj_kernel<<<dim3(192), blk, 0, stream>>>(query, Wq_b, nullptr, nullptr,
                                             bq, nullptr, Qb, 0);
  proj_kernel<<<dim3(192), blk, 0, stream>>>(key_, Wk_b, kginf, Wkg_b,
                                             bk, bkg, KST, 1);   // KS = K + KG
  proj_kernel<<<dim3(192), blk, 0, stream>>>(value, Wv_b, nullptr, nullptr,
                                             bv, nullptr, Vb, 0);

  // 3. raw masked scores into attn region of d_out (49152 waves)
  scores_kernel<<<dim3(6144), blk, 0, stream>>>(Qb, KST, mask, attnW);

  // 4. per-row softmax stats (49152 rows, 1 wave each)
  stats_kernel<<<dim3(6144), blk, 0, stream>>>(attnW, stats);

  // 5. normalize in place + attn @ V (1536 waves)
  normav_kernel<<<dim3(192), blk, 0, stream>>>(attnW, Vb, stats, AOb);

  // 6. output projection -> d_out f32
  out_kernel<<<dim3(192), blk, 0, stream>>>(AOb, Wo_b, bo, out);
}
